// SSIM_71141838291536
// MI455X (gfx1250) — matrix operand order repaired and checked by reading, not compiled, not run on hardware
//
#include <hip/hip_runtime.h>

typedef __attribute__((ext_vector_type(16))) _Float16 v16h;
typedef __attribute__((ext_vector_type(8)))  _Float16 v8h;
typedef __attribute__((ext_vector_type(8)))  float    v8f;

// Tile geometry: each workgroup produces a TY x TX block of conv outputs.
#define TY 64          // output rows per workgroup
#define TX 32          // output cols per workgroup
#define IN_ROWS 80     // padded input rows (74 = TY+10 used)
#define IN_STR  56     // input row stride in halves (48 cols used: 42=TX+10 data + zero pad); 16B-aligned, conflict-free
#define T_STR   88     // intermediate column stride in halves (80 rows); 16B-aligned, conflict-free
#define NSTR 5         // mu1, mu2, x^2, y^2, x*y streams
#define OUTW 502       // 512 - 11 + 1

__global__ __launch_bounds__(256) void ssim_fused(
    const float* __restrict__ img1, const float* __restrict__ img2,
    const float* __restrict__ win, float* __restrict__ partials)
{
  __shared__ __align__(16) _Float16 ldsIn[2 * IN_ROWS * IN_STR]; // f16 img1, img2 tiles
  __shared__ __align__(16) _Float16 ldsT[NSTR * TX * T_STR];     // horizontal results, column-major
  __shared__ float ldsW[32];                                     // [0..10]=hrow, [16..26]=vcol
  __shared__ float red[256];

  const int tid  = threadIdx.x;
  const int lane = tid & 31;
  const int wv   = tid >> 5;        // wave id 0..7
  const int lh   = lane >> 4;       // lane half
  const int ln   = lane & 15;
  const int tx = blockIdx.x, ty = blockIdx.y, nc = blockIdx.z;
  const int x0 = tx * TX, y0 = ty * TY;
  const long ibase = (long)nc * 512 * 512;

  // ---- phase 0: 1D separable weights to LDS, zero-fill input staging ----
  if (tid < 16) {
    float h = (tid < 11) ? win[55 + tid] : 0.f;           // hrow[j] = g5*g[j]/3
    float v = (tid < 11) ? win[tid * 11 + 5] / win[60] : 0.f; // vcol[i] = g[i]/g5
    ldsW[tid] = h;
    ldsW[16 + tid] = v;
  }
  for (int i = tid; i < 2 * IN_ROWS * IN_STR / 2; i += 256)
    ((unsigned int*)ldsIn)[i] = 0u;
  __syncthreads();

  // ---- build constant WMMA weight fragments (hi/lo f16 split) ----
  // Horizontal pass B (32x16 f16): B[K][x] = hrow[K-x]; lane: N=ln, half h -> K = h + 8*((h>=8)+lh).
  // Vertical pass   A (16x32 f16): A[y][K] = vcol[K-y]; lane: M=ln, half h -> K = h + 16*lh.
  union U16 { v16h v; v8h p[2]; _Float16 e[16]; };
  U16 bhH, bhL, vaH, vaL;
#pragma unroll
  for (int h = 0; h < 16; ++h) {
    int Kb = h + 8 * ((h >= 8) + lh);
    int db = Kb - ln;
    float wb = (db >= 0 && db < 11) ? ldsW[db] : 0.f;
    _Float16 hb = (_Float16)wb;
    bhH.e[h] = hb;
    bhL.e[h] = (_Float16)(wb - (float)hb);
    int Ka = h + 16 * lh;
    int da = Ka - ln;
    float wa = (da >= 0 && da < 11) ? ldsW[16 + da] : 0.f;
    _Float16 ha = (_Float16)wa;
    vaH.e[h] = ha;
    vaL.e[h] = (_Float16)(wa - (float)ha);
  }

  // ---- phase 1: stage clamped input tile as f16 ----
  _Float16* lX = ldsIn;
  _Float16* lY = ldsIn + IN_ROWS * IN_STR;
  for (int p = tid; p < 74 * 42; p += 256) {
    int r = p / 42, c = p - r * 42;
    int gy = min(y0 + r, 511);
    int gx = min(x0 + c, 511);
    float a = img1[ibase + gy * 512 + gx];
    float b = img2[ibase + gy * 512 + gx];
    lX[r * IN_STR + c] = (_Float16)a;
    lY[r * IN_STR + c] = (_Float16)b;
  }
  __syncthreads();

  // ---- phase 2: horizontal pass (data in A, banded hrow in B) ----
  // 5 row-tiles x 2 col-tiles = 10 spatial tiles; each: 5 streams x 2 WMMAs.
  for (int t = wv; t < 10; t += 8) {
    int rt = t >> 1, ct = t & 1;
    int rowOff = (rt * 16 + ln) * IN_STR;
    int kb = ct * 16 + 8 * lh;               // K start for A halves 0..7
    U16 ax, ay, axx, ayy, axy;
    ax.p[0] = *(const v8h*)&lX[rowOff + kb];
    ax.p[1] = *(const v8h*)&lX[rowOff + kb + 16];
    ay.p[0] = *(const v8h*)&lY[rowOff + kb];
    ay.p[1] = *(const v8h*)&lY[rowOff + kb + 16];
    axx.v = ax.v * ax.v;                     // v_pk_mul_f16
    ayy.v = ay.v * ay.v;
    axy.v = ax.v * ay.v;
    v16h A[NSTR] = { ax.v, ay.v, axx.v, ayy.v, axy.v };
    int col = ct * 16 + ln;
    int row = rt * 16 + 8 * lh;
#pragma unroll
    for (int s = 0; s < NSTR; ++s) {
      v8f c = {};
      c = __builtin_amdgcn_wmma_f32_16x16x32_f16(false, A[s], false, bhH.v, (short)0, c, false, false);
      c = __builtin_amdgcn_wmma_f32_16x16x32_f16(false, A[s], false, bhL.v, (short)0, c, false, false);
      v8h pk;
#pragma unroll
      for (int r = 0; r < 8; ++r) pk[r] = (_Float16)c[r];
      *(v8h*)&ldsT[(s * TX + col) * T_STR + row] = pk;   // column-major store
    }
  }
  __syncthreads();

  // ---- phase 3: vertical pass (banded vcol in A, data in B) + SSIM epilogue ----
  float ssum = 0.f;
  {
    int yt = wv >> 1, xt = wv & 1;           // 4 x 2 spatial tiles, one per wave
    int col = xt * 16 + ln;
    int kb  = yt * 16 + 16 * lh;             // B rows for this lane half
    v8f acc[NSTR];
#pragma unroll
    for (int s = 0; s < NSTR; ++s) {
      U16 b;
      const _Float16* base = &ldsT[(s * TX + col) * T_STR + kb];
      b.p[0] = *(const v8h*)(base);
      b.p[1] = *(const v8h*)(base + 8);
      v8f c = {};
      c = __builtin_amdgcn_wmma_f32_16x16x32_f16(false, vaH.v, false, b.v, (short)0, c, false, false);
      c = __builtin_amdgcn_wmma_f32_16x16x32_f16(false, vaL.v, false, b.v, (short)0, c, false, false);
      acc[s] = c;
    }
    int gx = x0 + xt * 16 + ln;
#pragma unroll
    for (int r = 0; r < 8; ++r) {
      int gy = y0 + yt * 16 + 8 * lh + r;
      if (gy < OUTW && gx < OUTW) {
        float mu1 = acc[0][r], mu2 = acc[1][r];
        float m11 = mu1 * mu1, m22 = mu2 * mu2, m12 = mu1 * mu2;
        float s1 = acc[2][r] - m11;
        float s2 = acc[3][r] - m22;
        float s12 = acc[4][r] - m12;
        float num = (2.f * m12) * (2.f * s12);
        float den = (m11 + m22) * (s1 + s2);
        ssum += num / den;
      }
    }
  }

  // ---- phase 4: deterministic block reduction -> partial ----
  red[tid] = ssum;
  __syncthreads();
  for (int s = 128; s > 0; s >>= 1) {
    if (tid < s) red[tid] += red[tid + s];
    __syncthreads();
  }
  if (tid == 0) partials[(nc * 8 + ty) * 16 + tx] = red[0];
}

__global__ __launch_bounds__(256) void ssim_reduce(
    const float* __restrict__ partials, float* __restrict__ out)
{
  __shared__ float red[256];
  int tid = threadIdx.x;
  float s = 0.f;
  for (int i = tid; i < 48 * 8 * 16; i += 256) s += partials[i];  // fixed order
  red[tid] = s;
  __syncthreads();
  for (int k = 128; k > 0; k >>= 1) {
    if (tid < k) red[tid] += red[tid + k];
    __syncthreads();
  }
  if (tid == 0) out[0] = red[0] / 12096192.0f;  // 48 * 502 * 502
}

extern "C" void kernel_launch(void* const* d_in, const int* in_sizes, int n_in,
                              void* d_out, int out_size, void* d_ws, size_t ws_size,
                              hipStream_t stream) {
  (void)in_sizes; (void)n_in; (void)out_size; (void)ws_size;
  const float* img1 = (const float*)d_in[0];
  const float* img2 = (const float*)d_in[1];
  const float* win  = (const float*)d_in[2];
  float* partials = (float*)d_ws;              // 6144 floats = 24 KB scratch
  dim3 grid(16, 8, 48);                        // x-tiles, y-tiles, N*C
  dim3 block(256);
  ssim_fused<<<grid, block, 0, stream>>>(img1, img2, win, partials);
  ssim_reduce<<<dim3(1), block, 0, stream>>>(partials, (float*)d_out);
}
